// EntNet_49005576847824
// MI455X (gfx1250) — matrix operand order repaired and checked by reading, not compiled, not run on hardware
//
#include <hip/hip_runtime.h>
#include <math.h>

// Problem constants (match reference)
#define B_ 32
#define S_ 256
#define L_ 64
#define D_ 100
#define M_ 20

typedef __attribute__((ext_vector_type(16))) _Float16 v16h;
typedef __attribute__((ext_vector_type(8)))  _Float16 v8h;
typedef __attribute__((ext_vector_type(8)))  float    v8f;

// ---------------------------------------------------------------------------
// Kernel 1: enc[b,s,d] = sum_l batch[b,s,l,d] * enc_mult[l,d]
// HBM-bound streaming reduction (~210 MB @ 23.3 TB/s); one block per (b,s),
// threads own d (consecutive threads -> consecutive addresses -> coalesced).
// ---------------------------------------------------------------------------
__global__ __launch_bounds__(128) void entnet_encode(
    const float* __restrict__ batch,
    const float* __restrict__ enc_mult,
    float* __restrict__ enc)
{
    const int bs = blockIdx.x;      // 0 .. B*S-1
    const int d  = threadIdx.x;     // 0 .. 127
    if (d >= D_) return;
    const float* src = batch + (size_t)bs * (L_ * D_) + d;
    float acc = 0.f;
#pragma unroll 8
    for (int l = 0; l < L_; ++l)
        acc = fmaf(src[(size_t)l * D_], enc_mult[l * D_ + d], acc);
    enc[(size_t)bs * D_ + d] = acc;
}

// ---------------------------------------------------------------------------
// Kernel 2: sequential EntNet scan. One block (256 threads = 8 waves) per
// batch element.
//   mem [20x100]: f32 master + f16 shadow in LDS, zero-padded to [32][128].
//   Per-step GEMM cand_pre = mem @ U^T on v_wmma_f32_16x16x32_f16
//     (M tiles 2, N tiles 7 -> waves 0..6, K padded 100->128 -> 4 k-steps).
//   B fragments (U^T) step-invariant -> built once, register-resident.
//   A fragments: two ds_load_b128 of the f16 shadow per fragment.
//   Wave 7 computes gates + W@s concurrently with the WMMA phase.
// ---------------------------------------------------------------------------
__global__ __launch_bounds__(256) void entnet_scan(
    const float* __restrict__ enc,
    const float* __restrict__ keys,
    const float* __restrict__ U,
    const float* __restrict__ V,
    const float* __restrict__ W,
    const float* __restrict__ prelu_a,
    float* __restrict__ out)
{
    __shared__ float    s_mem [32 * 128];    // f32 master, 16 KB
    __shared__ _Float16 s_memh[32 * 128];    // f16 shadow (A-frag source), 8 KB
    __shared__ float    s_keys[M_ * D_];     // 8 KB
    __shared__ float    s_keysV[M_ * D_];    // 8 KB (step-invariant keys @ V^T)
    __shared__ _Float16 s_W[D_ * 104];       // W f16, 104-stride (16B aligned,
                                             // zero tail), ~20.8 KB
    __shared__ float    s_vec[104];          // sentence vector, zero tail
    __shared__ float    s_ws[D_];            // W @ s
    __shared__ float    s_gate[M_];
    __shared__ float    s_rnorm[M_];

    const int b    = blockIdx.x;
    const int tid  = threadIdx.x;
    const int lane = tid & 31;
    const int wid  = tid >> 5;
    const float ap = prelu_a[0];

    // ---- prologue: zero padded buffers ----
    for (int i = tid; i < 32 * 128; i += 256) {
        s_mem[i]  = 0.f;
        s_memh[i] = (_Float16)0.f;
    }
    if (tid < 104) s_vec[tid] = 0.f;
    __syncthreads();

    // keys -> LDS, mem0 = keys (both copies); W -> LDS f16 with zero tail
    for (int i = tid; i < M_ * D_; i += 256) {
        int m = i / D_, d = i - m * D_;
        float k = keys[i];
        s_keys[i] = k;
        s_mem [m * 128 + d] = k;
        s_memh[m * 128 + d] = (_Float16)k;
    }
    for (int i = tid; i < D_ * 104; i += 256) {
        int e = i / 104, d = i - e * 104;
        s_W[i] = (d < D_) ? (_Float16)W[e * D_ + d] : (_Float16)0.f;
    }
    __syncthreads();

    // keysV[m,e] = sum_d keys[m,d] * V[e,d]  (step-invariant, one-time)
    for (int i = tid; i < M_ * D_; i += 256) {
        int m = i / D_, e = i - m * D_;
        float acc = 0.f;
        for (int d = 0; d < D_; ++d)
            acc = fmaf(s_keys[m * D_ + d], V[e * D_ + d], acc);
        s_keysV[i] = acc;
    }

    // ---- B fragments: B[k,n] = U[n,k] (= U^T), constant over all steps ----
    // Contiguous k order used here (k = 16h + s, same as A): VGPR v: lanes 0-15 hold K=2v,2v+1;
    // lanes 16-31 hold K=16+2v,16+2v+1; k-step kk adds 32.
    v16h bfrag[4];
    {
        const int n     = wid * 16 + (lane & 15);   // wid==7 -> n>=112 -> zeros
        const int khalf = (lane >> 4) << 4;         // 0 or 16
#pragma unroll
        for (int kk = 0; kk < 4; ++kk) {
            v16h bf;
#pragma unroll
            for (int v = 0; v < 8; ++v) {
                int k = kk * 32 + khalf + 2 * v;
                float b0 = (n < D_ && k     < D_) ? U[n * D_ + k    ] : 0.f;
                float b1 = (n < D_ && k + 1 < D_) ? U[n * D_ + k + 1] : 0.f;
                bf[2 * v]     = (_Float16)b0;
                bf[2 * v + 1] = (_Float16)b1;
            }
            bfrag[kk] = bf;
        }
    }
    __syncthreads();

    // ---- sequential scan over sentences ----
    for (int step = 0; step < S_; ++step) {
        // load sentence vector s, prefetch next step's (400B -> 4 lines)
        if (tid < D_) s_vec[tid] = enc[((size_t)b * S_ + step) * D_ + tid];
        if (step + 1 < S_ && tid < 4)
            __builtin_prefetch(&enc[((size_t)b * S_ + step + 1) * D_ + tid * 32], 0, 0);
        __syncthreads();

        // ===== phase A (overlapped) =====
        v8f c0 = {};
        v8f c1 = {};
        if (wid < 7) {
            // memU = mem @ U^T on the matrix unit. Wave-uniform branch keeps
            // EXEC all-1s at the WMMA (ISA requirement).
            const int mrow = lane & 15;
            const int h8   = (lane >> 4) << 4;   // 0 or 16 (halves)
#pragma unroll
            for (int kk = 0; kk < 4; ++kk) {
                // A in contiguous k order (k = 16h + s, same as B): VGPRs 0-3: K = kk*32 + h*16 + 0..7,
                // VGPRs 4-7: K = kk*32 + h*16 + 8..15 -> two b128 runs.
                const int k0 = kk * 32 + h8;
                v8h lo0 = *(const v8h*)&s_memh[ mrow       * 128 + k0];
                v8h hi0 = *(const v8h*)&s_memh[ mrow       * 128 + k0 + 8];
                v8h lo1 = *(const v8h*)&s_memh[(16 + mrow) * 128 + k0];
                v8h hi1 = *(const v8h*)&s_memh[(16 + mrow) * 128 + k0 + 8];
                v16h a0 = __builtin_shufflevector(lo0, hi0,
                    0,1,2,3,4,5,6,7,8,9,10,11,12,13,14,15);
                v16h a1 = __builtin_shufflevector(lo1, hi1,
                    0,1,2,3,4,5,6,7,8,9,10,11,12,13,14,15);
                c0 = __builtin_amdgcn_wmma_f32_16x16x32_f16(
                        false, a0, false, bfrag[kk], (short)0, c0, false, false);
                c1 = __builtin_amdgcn_wmma_f32_16x16x32_f16(
                        false, a1, false, bfrag[kk], (short)0, c1, false, false);
            }
        } else {
            // wave 7: gates + W@s, concurrent with the WMMA waves.
            // gate[m] = sigmoid(s.mem[m] + s.keys[m]), float4-vectorized.
            if (lane < M_) {
                const int m = lane;
                float mg = 0.f, kg = 0.f;
#pragma unroll 5
                for (int c = 0; c < 25; ++c) {
                    float4 sv = *(const float4*)&s_vec [c * 4];
                    float4 mv = *(const float4*)&s_mem [m * 128 + c * 4];
                    float4 kv = *(const float4*)&s_keys[m * D_  + c * 4];
                    mg = fmaf(sv.x, mv.x, fmaf(sv.y, mv.y,
                         fmaf(sv.z, mv.z, fmaf(sv.w, mv.w, mg))));
                    kg = fmaf(sv.x, kv.x, fmaf(sv.y, kv.y,
                         fmaf(sv.z, kv.z, fmaf(sv.w, kv.w, kg))));
                }
                float g = mg + kg;
                s_gate[m] = 1.f / (1.f + __expf(-g));
            }
            // ws[e] = sum_d W[e,d]*s[d], v8h chunks over zero-padded rows.
            for (int e = lane; e < D_; e += 32) {
                float acc = 0.f;
#pragma unroll
                for (int c = 0; c < 13; ++c) {
                    v8h wc = *(const v8h*)&s_W[e * 104 + c * 8];
#pragma unroll
                    for (int j = 0; j < 8; ++j)
                        acc = fmaf((float)wc[j], s_vec[c * 8 + j], acc);
                }
                s_ws[e] = acc;
            }
        }
        __syncthreads();   // ws/gates ready; all reads of old mem complete

        // ===== write-back: mem += prelu(memU + keysV + ws) * gate =====
        // C layout: VGPR r -> (M = r + 8*h, N = lane&15) within tile.
        // m0 = r + 8*h <= 15 < M_ always; m1 = 16 + r + 8*h valid iff h==0, r<4.
        if (wid < 7) {
            const int e = wid * 16 + (lane & 15);
            if (e < D_) {
                const int h8  = (lane >> 4) << 3;
                const float wse = s_ws[e];
#pragma unroll
                for (int r = 0; r < 8; ++r) {
                    int m0 = r + h8;
                    float cv   = c0[r] + s_keysV[m0 * D_ + e] + wse;
                    float cand = cv >= 0.f ? cv : ap * cv;
                    s_mem[m0 * 128 + e] += cand * s_gate[m0];
                }
                if ((lane >> 4) == 0) {
#pragma unroll
                    for (int r = 0; r < 4; ++r) {
                        int m1 = 16 + r;
                        float cv   = c1[r] + s_keysV[m1 * D_ + e] + wse;
                        float cand = cv >= 0.f ? cv : ap * cv;
                        s_mem[m1 * 128 + e] += cand * s_gate[m1];
                    }
                }
            }
        }
        __syncthreads();

        // ===== L2 row norms (float4-vectorized) =====
        if (tid < M_) {
            float ss = 0.f;
#pragma unroll 5
            for (int c = 0; c < 25; ++c) {
                float4 x = *(const float4*)&s_mem[tid * 128 + c * 4];
                ss = fmaf(x.x, x.x, fmaf(x.y, x.y,
                     fmaf(x.z, x.z, fmaf(x.w, x.w, ss))));
            }
            s_rnorm[tid] = rsqrtf(ss);
        }
        __syncthreads();

        // ===== scale f32 master + refresh f16 shadow (float2 / half2) =====
        for (int i = tid; i < (M_ * D_) / 2; i += 256) {
            int m  = i / 50;
            int e2 = (i - m * 50) * 2;
            float2 v = *(const float2*)&s_mem[m * 128 + e2];
            float  rn = s_rnorm[m];
            v.x *= rn; v.y *= rn;
            *(float2*)&s_mem[m * 128 + e2] = v;
            _Float16 h2[2] = { (_Float16)v.x, (_Float16)v.y };
            *(__attribute__((ext_vector_type(2))) _Float16*)&s_memh[m * 128 + e2] =
                *(__attribute__((ext_vector_type(2))) _Float16*)h2;
        }
        __syncthreads();
    }

    // ---- write final memories ----
    for (int i = tid; i < M_ * D_; i += 256) {
        int m = i / D_, e = i - m * D_;
        out[((size_t)b * M_ + m) * D_ + e] = s_mem[m * 128 + e];
    }
}

// ---------------------------------------------------------------------------
extern "C" void kernel_launch(void* const* d_in, const int* in_sizes, int n_in,
                              void* d_out, int out_size, void* d_ws, size_t ws_size,
                              hipStream_t stream)
{
    const float* batch    = (const float*)d_in[0];  // [B,S,L,D]
    const float* enc_mult = (const float*)d_in[1];  // [L,D]
    const float* keys     = (const float*)d_in[2];  // [M,D]
    const float* U        = (const float*)d_in[3];  // [D,D]
    const float* V        = (const float*)d_in[4];  // [D,D]
    const float* W        = (const float*)d_in[5];  // [D,D]
    const float* prelu_a  = (const float*)d_in[6];  // scalar
    float* out = (float*)d_out;                     // [B,M,D]
    float* enc = (float*)d_ws;                      // scratch: [B,S,D] floats

    entnet_encode<<<B_ * S_, 128, 0, stream>>>(batch, enc_mult, enc);
    entnet_scan<<<B_, 256, 0, stream>>>(enc, keys, U, V, W, prelu_a, out);
}
